// MultiHeadAttention_2869038154480
// MI455X (gfx1250) — matrix operand order repaired and checked by reading, not compiled, not run on hardware
//
#include <hip/hip_runtime.h>
#include <stdint.h>

// ---------------------------------------------------------------------------
// MI455X (gfx1250) fused multi-head attention:
//   bf16 WMMA (v_wmma_f32_16x16x32_bf16) with fp32 accumulation.
//   Q' = q@wq+bq  -> ws as bf16 [B][H][S][64]
//   K' = k@wk+bk  -> ws as bf16 [B][H][S][64]
//   V' = v@wv+bv  -> ws as bf16 [B][H][64][S]   (transposed for PV B-frags)
//   flash attention (K/V tiles staged via GLOBAL_LOAD_ASYNC_TO_LDS_B128)
//   out = attn@wo+bo -> d_out fp32
// ---------------------------------------------------------------------------

constexpr int D_MODEL   = 1024;
constexpr int NUM_HEADS = 16;
constexpr int DEPTH     = 64;
constexpr int Bb        = 2;
constexpr int Ss        = 2048;

constexpr int BM = 128, BN = 128, BK = 32;

typedef __attribute__((ext_vector_type(16))) __bf16          v16bf;
typedef __attribute__((ext_vector_type(8)))  float           v8f;
typedef __attribute__((ext_vector_type(8)))  unsigned short  u16x8;

// ---- gfx1250 async global->LDS copy (ASYNCcnt path), with fallback --------
#if defined(__has_builtin)
#  if __has_builtin(__builtin_amdgcn_global_load_async_to_lds_b128) && \
      __has_builtin(__builtin_amdgcn_s_wait_asynccnt)
#    define HAVE_ASYNC_LDS 1
#  endif
#endif
#ifndef HAVE_ASYNC_LDS
#  define HAVE_ASYNC_LDS 0
#endif

typedef int v4i __attribute__((vector_size(16)));            // matches builtin proto
typedef __attribute__((address_space(1))) v4i* g4ptr_t;      // global
typedef __attribute__((address_space(3))) v4i* l4ptr_t;      // LDS

static __device__ __forceinline__ void async_copy16(const void* gsrc, void* lds_dst) {
#if HAVE_ASYNC_LDS
  g4ptr_t g = (g4ptr_t)(unsigned long long)(uintptr_t)gsrc;
  l4ptr_t l = (l4ptr_t)(unsigned)(uintptr_t)lds_dst;   // low 32 bits = LDS offset
  __builtin_amdgcn_global_load_async_to_lds_b128(g, l, 0, 0);
#else
  *(u16x8*)lds_dst = *(const u16x8*)gsrc;
#endif
}
static __device__ __forceinline__ void async_fence() {
#if HAVE_ASYNC_LDS
  __builtin_amdgcn_s_wait_asynccnt(0);
#endif
}

static __device__ __forceinline__ __bf16 f2bf(float f) {
  unsigned u = __builtin_bit_cast(unsigned, f);
  unsigned r = u + 0x7FFFu + ((u >> 16) & 1u);   // round-to-nearest-even
  unsigned short h = (unsigned short)(r >> 16);
  return __builtin_bit_cast(__bf16, h);
}
static __device__ __forceinline__ __bf16 to_bf16(float f)  { return f2bf(f); }
static __device__ __forceinline__ __bf16 to_bf16(__bf16 h) { return h; }

// Load a 16-element bf16 fragment as two 16-byte runs (b128-friendly).
static __device__ __forceinline__ v16bf load_frag(const __bf16* p0, const __bf16* p1) {
  union { u16x8 h[2]; v16bf v; } u;
  u.h[0] = *(const u16x8*)p0;
  u.h[1] = *(const u16x8*)p1;
  return u.v;
}

// ---------------------------------------------------------------------------
// GEMM: C[M,N] = A[M,K] @ W[K,N] + bias   (A: fp32 or bf16; W/bias fp32)
// OUT_MODE 0: bf16 out, head-split   [B][H][S][64]
// OUT_MODE 1: bf16 out, head-split-T [B][H][64][S]
// OUT_MODE 2: fp32 out, row-major    [M][N]
// Block: 256 threads = 8 waves in 2x4; wave tile 64x32 (4x2 WMMA tiles).
// ---------------------------------------------------------------------------
template <typename TIn, int OUT_MODE>
__global__ __launch_bounds__(256) void gemm_bias_wmma(
    const TIn* __restrict__ A, const float* __restrict__ W,
    const float* __restrict__ bias, void* __restrict__ out,
    int M, int K, int N)
{
  __shared__ alignas(16) __bf16 sA [BM][BK];   // A tile, row-major
  __shared__ alignas(16) __bf16 sBt[BN][BK];   // W tile, transposed [n][k]

  const int tid  = threadIdx.x;
  const int lane = tid & 31, wave = tid >> 5;
  const int half = lane >> 4, l16 = lane & 15;
  const int wm   = wave >> 2, wn = wave & 3;
  const int m0   = blockIdx.y * BM, n0 = blockIdx.x * BN;

  v8f acc[4][2];
  #pragma unroll
  for (int t = 0; t < 4; ++t)
    #pragma unroll
    for (int u = 0; u < 2; ++u) acc[t][u] = v8f{};

  for (int k0 = 0; k0 < K; k0 += BK) {
    // ---- stage A tile (128x32 bf16) ----
    if constexpr (sizeof(TIn) == 2) {
      // already bf16: raw async copy, 512x16B chunks, 2 per thread
      #pragma unroll
      for (int i = 0; i < 2; ++i) {
        const int c = tid * 2 + i;
        const int r = c >> 2, off = (c & 3) * 8;
        async_copy16(A + (size_t)(m0 + r) * K + k0 + off, &sA[r][off]);
      }
    } else {
      // fp32: load + convert on the VALU
      const int c = (tid & 7) * 4;
      #pragma unroll
      for (int p = 0; p < 4; ++p) {
        const int r = p * 32 + (tid >> 3);
        const TIn* src = A + (size_t)(m0 + r) * K + k0 + c;
        #pragma unroll
        for (int i = 0; i < 4; ++i) sA[r][c + i] = to_bf16(src[i]);
      }
    }
    // ---- stage W^T tile (32x128 -> [n][k]), fp32 -> bf16 ----
    {
      const int kk = tid >> 3;
      const int nn = (tid & 7) * 16;
      const float* src = W + (size_t)(k0 + kk) * N + n0 + nn;
      #pragma unroll
      for (int i = 0; i < 16; ++i) sBt[nn + i][kk] = f2bf(src[i]);
    }
    if (k0 + BK < K) {  // global_prefetch_b8 for next K-slab
      __builtin_prefetch(A + (size_t)(m0 + (tid >> 3)) * K + k0 + BK, 0, 1);
      __builtin_prefetch(W + (size_t)(k0 + BK + (tid >> 3)) * N + n0, 0, 1);
    }
    async_fence();
    __syncthreads();

    // ---- fragments (ISA 7.12.2 layouts) ----
    v16bf af[4], bfr[2];
    #pragma unroll
    for (int t = 0; t < 4; ++t) {
      // A-frag: lane row = l16; K runs {h*8 .. h*8+7} and {h*8+16 .. h*8+23}
      const __bf16* p = &sA[wm * 64 + t * 16 + l16][half * 8];
      af[t] = load_frag(p, p + 16);
    }
    #pragma unroll
    for (int u = 0; u < 2; ++u) {
      // B-frag: lane col n = l16; K runs {h*8 .. h*8+7} and {h*8+16 .. h*8+23}
      const __bf16* p = &sBt[wn * 32 + u * 16 + l16][half * 8];
      bfr[u] = load_frag(p, p + 16);
    }
    #pragma unroll
    for (int t = 0; t < 4; ++t)
      #pragma unroll
      for (int u = 0; u < 2; ++u)
        acc[t][u] = __builtin_amdgcn_wmma_f32_16x16x32_bf16(
            false, af[t], false, bfr[u], (short)0, acc[t][u], false, false);
    __syncthreads();
  }

  // ---- epilogue: bias + scatter per layout ----
  #pragma unroll
  for (int t = 0; t < 4; ++t) {
    #pragma unroll
    for (int u = 0; u < 2; ++u) {
      const int n = n0 + wn * 32 + u * 16 + l16;
      const float bia = bias[n];
      #pragma unroll
      for (int r = 0; r < 8; ++r) {
        const int m = m0 + wm * 64 + t * 16 + r + half * 8;
        const float val = acc[t][u][r] + bia;
        if (OUT_MODE == 2) {
          ((float*)out)[(size_t)m * N + n] = val;
        } else {
          const int b = m / Ss, s = m % Ss;
          const int h = n >> 6, d = n & 63;
          size_t idx;
          if (OUT_MODE == 0)
            idx = (((size_t)b * NUM_HEADS + h) * Ss + s) * DEPTH + d;
          else
            idx = (((size_t)b * NUM_HEADS + h) * DEPTH + d) * Ss + s;
          ((__bf16*)out)[idx] = f2bf(val);
        }
      }
    }
  }
}

// ---------------------------------------------------------------------------
// Flash attention: block = 4 waves, each wave owns 16 queries of one (b,h).
// K/V tiles (64 keys) staged into LDS via async-to-LDS, shared by all waves.
// Softmax fp32 with shfl-xor row reductions; P bounced through LDS to turn
// the C-layout logits into A-layout fragments.
// ---------------------------------------------------------------------------
__global__ __launch_bounds__(128) void flash_attn_wmma(
    const __bf16* __restrict__ Qh, const __bf16* __restrict__ Kh,
    const __bf16* __restrict__ Vt, __bf16* __restrict__ Oc)
{
  __shared__ alignas(16) __bf16 sK[64][64];      // [key][depth]  8 KB
  __shared__ alignas(16) __bf16 sV[64][64];      // [depth][key]  8 KB
  __shared__ alignas(16) __bf16 sP[4][16][64];   // per-wave P    8 KB

  const int tid  = threadIdx.x;
  const int lane = tid & 31, wave = tid >> 5;
  const int half = lane >> 4, l16 = lane & 15;
  const int h = blockIdx.y, b = blockIdx.z;
  const int q0 = blockIdx.x * 64 + wave * 16;

  const __bf16* qb = Qh + (((size_t)b * NUM_HEADS + h) * Ss) * DEPTH;
  const __bf16* kb = Kh + (((size_t)b * NUM_HEADS + h) * Ss) * DEPTH;
  const __bf16* vb = Vt + (((size_t)b * NUM_HEADS + h) * DEPTH) * Ss;

  // resident Q A-fragments (depth 64 = 2 chunks of 32)
  v16bf aq[2];
  #pragma unroll
  for (int c = 0; c < 2; ++c) {
    const __bf16* p = qb + (size_t)(q0 + l16) * DEPTH + c * 32 + half * 8;
    aq[c] = load_frag(p, p + 16);
  }

  v8f o[4];
  #pragma unroll
  for (int t = 0; t < 4; ++t) o[t] = v8f{};
  float mrow[8], lrow[8];
  #pragma unroll
  for (int r = 0; r < 8; ++r) { mrow[r] = -1e30f; lrow[r] = 0.f; }

  for (int kv0 = 0; kv0 < Ss; kv0 += 64) {
    // ---- async-stage K tile [64 keys][64 d] and V tile [64 d][64 keys] ----
    #pragma unroll
    for (int i = 0; i < 4; ++i) {
      const int c   = tid * 4 + i;        // 0..511, 16B chunks
      const int row = c >> 3;
      const int off = (c & 7) * 8;
      async_copy16(kb + (size_t)(kv0 + row) * DEPTH + off, &sK[row][off]);
      async_copy16(vb + (size_t)row * Ss + kv0 + off,      &sV[row][off]);
    }
    async_fence();
    __syncthreads();

    // ---- logits: 4 tiles of 16 keys, contraction over depth 64 ----
    float sv[4][8];
    #pragma unroll
    for (int t = 0; t < 4; ++t) {
      v8f s = v8f{};
      #pragma unroll
      for (int c = 0; c < 2; ++c) {
        // B-frag of K^T: lane col = key, depth runs {h*8..h*8+7} and {h*8+16..h*8+23}
        const __bf16* p = &sK[t * 16 + l16][c * 32 + half * 8];
        v16bf bk = load_frag(p, p + 16);
        s = __builtin_amdgcn_wmma_f32_16x16x32_bf16(
            false, aq[c], false, bk, (short)0, s, false, false);
      }
      #pragma unroll
      for (int r = 0; r < 8; ++r) sv[t][r] = s[r] * 0.125f;  // 1/sqrt(64)
    }
    // ---- online softmax: row max / rescale ----
    #pragma unroll
    for (int r = 0; r < 8; ++r) {
      float mx = sv[0][r];
      #pragma unroll
      for (int t = 1; t < 4; ++t) mx = fmaxf(mx, sv[t][r]);
      #pragma unroll
      for (int off = 1; off < 16; off <<= 1)
        mx = fmaxf(mx, __shfl_xor(mx, off, 32));
      const float mnew  = fmaxf(mrow[r], mx);
      const float alpha = __expf(mrow[r] - mnew);
      mrow[r] = mnew;
      lrow[r] *= alpha;
      #pragma unroll
      for (int t = 0; t < 4; ++t) o[t][r] *= alpha;
    }
    // ---- P = exp(s-m): stash bf16 in LDS (C-layout -> A-layout bounce) ----
    #pragma unroll
    for (int r = 0; r < 8; ++r) {
      float rs = 0.f;
      #pragma unroll
      for (int t = 0; t < 4; ++t) {
        const float pv = __expf(sv[t][r] - mrow[r]);
        rs += pv;
        sP[wave][r + half * 8][t * 16 + l16] = f2bf(pv);
      }
      #pragma unroll
      for (int off = 1; off < 16; off <<= 1) rs += __shfl_xor(rs, off, 32);
      lrow[r] += rs;
    }
    __syncthreads();
    // ---- O += P @ V ----
    v16bf ap[2];
    #pragma unroll
    for (int c = 0; c < 2; ++c) {
      const __bf16* p = &sP[wave][l16][c * 32 + half * 8];
      ap[c] = load_frag(p, p + 16);
    }
    #pragma unroll
    for (int t = 0; t < 4; ++t) {
      #pragma unroll
      for (int c = 0; c < 2; ++c) {
        // B-frag of V^T: lane col = depth d, key runs {h*8..h*8+7} and {h*8+16..h*8+23}
        const __bf16* p = &sV[t * 16 + l16][c * 32 + half * 8];
        v16bf bvf = load_frag(p, p + 16);
        o[t] = __builtin_amdgcn_wmma_f32_16x16x32_bf16(
            false, ap[c], false, bvf, (short)0, o[t], false, false);
      }
    }
    __syncthreads();
  }

  // ---- normalize, merge heads: Oc[b][s][h*64+d] bf16 ----
  #pragma unroll
  for (int t = 0; t < 4; ++t) {
    #pragma unroll
    for (int r = 0; r < 8; ++r) {
      const int q = q0 + r + half * 8;
      const int d = t * 16 + l16;
      const float val = o[t][r] / lrow[r];
      Oc[((size_t)b * Ss + q) * D_MODEL + h * DEPTH + d] = f2bf(val);
    }
  }
}

// ---------------------------------------------------------------------------
extern "C" void kernel_launch(void* const* d_in, const int* in_sizes, int n_in,
                              void* d_out, int out_size, void* d_ws, size_t ws_size,
                              hipStream_t stream) {
  const float* v  = (const float*)d_in[0];
  const float* k  = (const float*)d_in[1];
  const float* q  = (const float*)d_in[2];
  const float* wq = (const float*)d_in[3];
  const float* bq = (const float*)d_in[4];
  const float* wk = (const float*)d_in[5];
  const float* bk = (const float*)d_in[6];
  const float* wv = (const float*)d_in[7];
  const float* bv = (const float*)d_in[8];
  const float* wo = (const float*)d_in[9];
  const float* bo = (const float*)d_in[10];

  char* ws = (char*)d_ws;
  const size_t seg = (size_t)Bb * NUM_HEADS * Ss * DEPTH * sizeof(__bf16); // 8 MB
  __bf16* Qh = (__bf16*)(ws);
  __bf16* Kh = (__bf16*)(ws + seg);
  __bf16* Vt = (__bf16*)(ws + 2 * seg);
  __bf16* Oc = (__bf16*)(ws + 3 * seg);

  const int M = Bb * Ss, N = D_MODEL, K = D_MODEL;
  dim3 gg(N / BN, M / BM);

  gemm_bias_wmma<float, 0><<<gg, 256, 0, stream>>>(q, wq, bq, Qh, M, K, N);
  gemm_bias_wmma<float, 0><<<gg, 256, 0, stream>>>(k, wk, bk, Kh, M, K, N);
  gemm_bias_wmma<float, 1><<<gg, 256, 0, stream>>>(v, wv, bv, Vt, M, K, N);

  flash_attn_wmma<<<dim3(Ss / 64, NUM_HEADS, Bb), 128, 0, stream>>>(Qh, Kh, Vt, Oc);

  gemm_bias_wmma<__bf16, 2><<<gg, 256, 0, stream>>>(Oc, wo, bo, (float*)d_out, M, K, N);
}
